// GraphSAGE_GCN_45913200394644
// MI455X (gfx1250) — matrix operand order repaired and checked by reading, not compiled, not run on hardware
//
#include <hip/hip_runtime.h>
#include <stdint.h>

#define NODES 50000
#define NPAD  50048          // 391 * 128
#define EDGES 800000
#define F_IN  128
#define HID   256
#define EPSB  1e-5f

typedef __attribute__((ext_vector_type(16))) __bf16    v16bf;
typedef __attribute__((ext_vector_type(8)))  float     v8f;
typedef __attribute__((ext_vector_type(8)))  uint32_t  v8u;

__device__ __forceinline__ uint16_t f2bf(float f) {
  union { float f; uint32_t u; } v; v.f = f;
  uint32_t u = v.u;
  uint32_t r = u + 0x7FFFu + ((u >> 16) & 1u);   // round to nearest even
  return (uint16_t)(r >> 16);
}

// ---------------- degree ----------------
__global__ void k_deg(const long long* __restrict__ ei, float* __restrict__ deg) {
  int e = blockIdx.x * blockDim.x + threadIdx.x;
  if (e < EDGES) atomicAdd(deg + (int)ei[EDGES + e], 1.0f);
}

__global__ void k_deginv(const float* __restrict__ deg, float* __restrict__ dinv) {
  int i = blockIdx.x * blockDim.x + threadIdx.x;
  if (i < NPAD) dinv[i] = (i < NODES) ? 1.0f / fmaxf(deg[i], 1.0f) : 0.0f;
}

// ---------------- edge scatter: agg[dst] += x[src] ----------------
__global__ void k_scatter(const long long* __restrict__ ei, const float* __restrict__ x,
                          float* __restrict__ agg, int cshift, int cmask) {
  long long gid = (long long)blockIdx.x * blockDim.x + threadIdx.x;
  if (gid >= ((long long)EDGES << cshift)) return;
  int e = (int)(gid >> cshift);
  int c = (int)gid & cmask;
  int din = (cmask + 1) << 2;
  int s = (int)ei[e];
  int d = (int)ei[EDGES + e];
  float4 v = *(const float4*)(x + (size_t)s * din + (c << 2));
  float* p = agg + (size_t)d * din + (c << 2);
  atomicAdd(p + 0, v.x); atomicAdd(p + 1, v.y);
  atomicAdd(p + 2, v.z); atomicAdd(p + 3, v.w);
}

// ---------------- build bf16 A = [agg * deg^-1 | x], zero-padded rows ----------------
__global__ void k_build_a(const float* __restrict__ agg, const float* __restrict__ x,
                          const float* __restrict__ dinv, uint16_t* __restrict__ Ab,
                          int din, int kshift, int kmask) {
  long long gid = (long long)blockIdx.x * blockDim.x + threadIdx.x;
  if (gid >= ((long long)NPAD << kshift)) return;
  int row = (int)(gid >> kshift);
  int k   = (int)gid & kmask;
  float val = 0.0f;
  if (row < NODES) {
    val = (k < din) ? agg[(size_t)row * din + k] * dinv[row]
                    : x[(size_t)row * din + (k - din)];
  }
  Ab[gid] = f2bf(val);
}

// ---------------- build bf16 Bt[n][k] = W[k][n], W = [Wn ; Wr] ----------------
__global__ void k_build_b(const float* __restrict__ Wn, const float* __restrict__ Wr,
                          uint16_t* __restrict__ Bt, int din, int kshift, int kmask) {
  int gid = blockIdx.x * blockDim.x + threadIdx.x;
  if (gid >= (HID << kshift)) return;
  int n = gid >> kshift;
  int k = gid & kmask;
  float w = (k < din) ? Wn[(size_t)k * HID + n] : Wr[(size_t)(k - din) * HID + n];
  Bt[gid] = f2bf(w);
}

// ---------------- WMMA GEMM: H[NPAD x 256] = Ab[NPAD x Kd] * Bt^T ----------------
// 8 waves: 4(M) x 2(N); block tile 128x128; wave tile 32x64 (2x4 WMMA tiles)
__global__ __launch_bounds__(256) void k_gemm(const uint16_t* __restrict__ Ab,
                                              const uint16_t* __restrict__ Bt,
                                              float* __restrict__ H, int Kd) {
  const int tid  = threadIdx.x;
  const int lane = tid & 31;
  const int wave = tid >> 5;
  const int half = lane >> 4;     // 0: lanes 0-15, 1: lanes 16-31
  const int r16  = lane & 15;
  const int wm   = wave >> 1;     // 0..3
  const int wn   = wave & 1;      // 0..1
  const int m0   = blockIdx.x * 128 + wm * 32;
  const int n0   = blockIdx.y * 128 + wn * 64;

  v8f acc[2][4] = {};

  // A fragment (16x32 bf16): lane group 0 holds K[0..7]+[16..23], group 1 K[8..15]+[24..31]
  const uint16_t* a0 = Ab + (size_t)(m0 + r16) * Kd + half * 8;
  const uint16_t* a1 = a0 + (size_t)16 * Kd;
  // B fragment (32x16 bf16): lane = column, group 0 K[0..7]+[16..23], group 1 K[8..15]+[24..31]
  const uint16_t* bb = Bt + (size_t)(n0 + r16) * Kd + half * 8;

  for (int k0 = 0; k0 < Kd; k0 += 32) {
    v16bf af[2], bf[4];
    {
      uint4 lo = *(const uint4*)(a0 + k0);
      uint4 hi = *(const uint4*)(a0 + k0 + 16);
      v8u bits = {lo.x, lo.y, lo.z, lo.w, hi.x, hi.y, hi.z, hi.w};
      af[0] = __builtin_bit_cast(v16bf, bits);
    }
    {
      uint4 lo = *(const uint4*)(a1 + k0);
      uint4 hi = *(const uint4*)(a1 + k0 + 16);
      v8u bits = {lo.x, lo.y, lo.z, lo.w, hi.x, hi.y, hi.z, hi.w};
      af[1] = __builtin_bit_cast(v16bf, bits);
    }
#pragma unroll
    for (int nt = 0; nt < 4; ++nt) {
      uint4 lo = *(const uint4*)(bb + (size_t)nt * 16 * Kd + k0);
      uint4 hi = *(const uint4*)(bb + (size_t)nt * 16 * Kd + k0 + 16);
      v8u bits = {lo.x, lo.y, lo.z, lo.w, hi.x, hi.y, hi.z, hi.w};
      bf[nt] = __builtin_bit_cast(v16bf, bits);
    }
#pragma unroll
    for (int mt = 0; mt < 2; ++mt)
#pragma unroll
      for (int nt = 0; nt < 4; ++nt)
        acc[mt][nt] = __builtin_amdgcn_wmma_f32_16x16x32_bf16(
            false, af[mt], false, bf[nt], (short)0, acc[mt][nt], false, false);
  }

  // C/D layout: lane -> col = n&15; VGPR j -> row j + 8*half
#pragma unroll
  for (int mt = 0; mt < 2; ++mt) {
    int rowb = m0 + mt * 16 + half * 8;
#pragma unroll
    for (int j = 0; j < 8; ++j) {
      int row = rowb + j;
      if (row < NODES) {
        float* hp = H + (size_t)row * HID + n0 + r16;
#pragma unroll
        for (int nt = 0; nt < 4; ++nt) hp[nt * 16] = acc[mt][nt][j];
      }
    }
  }
}

// ---------------- column sums for BN ----------------
__global__ void k_stats(const float* __restrict__ H, float* __restrict__ sums,
                        float* __restrict__ sumsq) {
  int c  = threadIdx.x;           // 0..255 = column
  int r0 = blockIdx.x * 256;
  int rend = r0 + 256; if (rend > NODES) rend = NODES;
  float s = 0.0f, s2 = 0.0f;
  for (int r = r0; r < rend; ++r) {
    float h = H[(size_t)r * HID + c];
    s += h; s2 += h * h;
  }
  atomicAdd(sums + c, s);
  atomicAdd(sumsq + c, s2);
}

__global__ void k_bnparams(const float* __restrict__ sums, const float* __restrict__ sumsq,
                           const float* __restrict__ g, const float* __restrict__ b,
                           float* __restrict__ scale, float* __restrict__ shiftv) {
  int c = threadIdx.x;
  float mu  = sums[c]  * (1.0f / NODES);
  float var = sumsq[c] * (1.0f / NODES) - mu * mu;
  float rs  = rsqrtf(var + EPSB);
  float sc  = g[c] * rs;
  scale[c]  = sc;
  shiftv[c] = b[c] - mu * sc;
}

__global__ void k_bnelu(const float* __restrict__ H, const float* __restrict__ scale,
                        const float* __restrict__ shiftv, float* __restrict__ out) {
  long long gid = (long long)blockIdx.x * blockDim.x + threadIdx.x;
  if (gid >= (long long)NODES * HID) return;
  int c = (int)gid & (HID - 1);
  float y = H[gid] * scale[c] + shiftv[c];
  out[gid] = (y > 0.0f) ? y : expm1f(y);
}

// ---------------- host driver ----------------
extern "C" void kernel_launch(void* const* d_in, const int* in_sizes, int n_in,
                              void* d_out, int out_size, void* d_ws, size_t ws_size,
                              hipStream_t stream) {
  const float*     x0 = (const float*)d_in[0];
  const long long* ei = (const long long*)d_in[1];

  char* ws = (char*)d_ws;
  size_t off = 0;
  auto carve = [&](size_t bytes) -> char* {
    char* p = ws + off;
    off = (off + bytes + 255) & ~(size_t)255;
    return p;
  };
  float*    dinv  = (float*)carve((size_t)NPAD * 4);
  uint16_t* Ab    = (uint16_t*)carve((size_t)NPAD * 512 * 2);
  uint16_t* Bt    = (uint16_t*)carve((size_t)HID * 512 * 2);
  float*    scr   = (float*)carve((size_t)NPAD * HID * 4);  // agg, then h_pre
  float*    h0    = (float*)carve((size_t)NPAD * HID * 4);
  float*    h1    = (float*)carve((size_t)NPAD * HID * 4);
  float*    sums  = (float*)carve(HID * 4);
  float*    sumsq = (float*)carve(HID * 4);
  float*    scale = (float*)carve(HID * 4);
  float*    shftv = (float*)carve(HID * 4);

  // degrees (scr temporarily holds deg)
  hipMemsetAsync(scr, 0, (size_t)NODES * 4, stream);
  k_deg<<<(EDGES + 255) / 256, 256, 0, stream>>>(ei, scr);
  k_deginv<<<(NPAD + 255) / 256, 256, 0, stream>>>(scr, dinv);

  const float* lin[3]  = { x0, h0, h1 };
  float*       lout[3] = { h0, h1, (float*)d_out };

  for (int l = 0; l < 3; ++l) {
    const int din    = l ? HID : F_IN;
    const int Kd     = 2 * din;
    const int cshift = (din == F_IN) ? 5 : 6;       // din/4 = 32 or 64
    const int kshift = (Kd == 256) ? 8 : 9;
    const float* Wn = (const float*)d_in[2 + 5 * l + 0];
    const float* Wr = (const float*)d_in[2 + 5 * l + 2];
    const float* g  = (const float*)d_in[2 + 5 * l + 3];
    const float* b  = (const float*)d_in[2 + 5 * l + 4];
    const float* xin = lin[l];

    hipMemsetAsync(scr, 0, (size_t)NODES * din * 4, stream);
    long long sc_total = (long long)EDGES << cshift;
    k_scatter<<<(unsigned)((sc_total + 255) / 256), 256, 0, stream>>>(
        ei, xin, scr, cshift, (1 << cshift) - 1);

    long long a_total = (long long)NPAD << kshift;
    k_build_a<<<(unsigned)((a_total + 255) / 256), 256, 0, stream>>>(
        scr, xin, dinv, Ab, din, kshift, (1 << kshift) - 1);
    k_build_b<<<((HID << kshift) + 255) / 256, 256, 0, stream>>>(
        Wn, Wr, Bt, din, kshift, (1 << kshift) - 1);

    dim3 gg(NPAD / 128, HID / 128);
    k_gemm<<<gg, 256, 0, stream>>>(Ab, Bt, scr, Kd);   // h_pre overwrites agg

    hipMemsetAsync(sums, 0, HID * 4, stream);
    hipMemsetAsync(sumsq, 0, HID * 4, stream);
    k_stats<<<(NODES + 255) / 256, 256, 0, stream>>>(scr, sums, sumsq);
    k_bnparams<<<1, HID, 0, stream>>>(sums, sumsq, g, b, scale, shftv);

    long long ne = (long long)NODES * HID;
    k_bnelu<<<(unsigned)((ne + 255) / 256), 256, 0, stream>>>(scr, scale, shftv, lout[l]);
  }
}
